// CoocOpModel_21036749816258
// MI455X (gfx1250) — matrix operand order repaired and checked by reading, not compiled, not run on hardware
//
#include <hip/hip_runtime.h>

#define DVEC 128
#define NTOK 4096                 // B*S = 4*1024
#define KTOT (DVEC * DVEC)        // 16384
#define NCHUNK (KTOT / 32)        // 512 K-chunks of 32
#define PACK_BYTES ((size_t)NCHUNK * 8 * 32 * 32)  // 4 MiB bf16 B-fragment buffer

typedef __attribute__((ext_vector_type(16))) __bf16        v16bf;
typedef __attribute__((ext_vector_type(16))) float         v16f;
typedef __attribute__((ext_vector_type(8)))  float         v8f;
typedef __attribute__((ext_vector_type(8)))  unsigned int  v8u;

__device__ __forceinline__ unsigned f2bf(float x) {
  union { float f; unsigned u; } v; v.f = x;
  unsigned r = v.u + 0x7FFFu + ((v.u >> 16) & 1u);   // round-to-nearest-even
  return r >> 16;
}

// ---------------------------------------------------------------------------
// Kernel 1: pack cooccurrences [i][j][z] fp32 -> bf16 in WMMA B-fragment
// layout. Fragment (c = k-chunk, n = z-tile, lane): z = n*16 + lane%16,
// K rows = c*32 + (lane<16 ? 0..15 : 16..31), two bf16 per dword (K even low).
// 131072 threads, one 32-byte fragment slice per thread.
// ---------------------------------------------------------------------------
__global__ void cooc_pack_kernel(const float* __restrict__ C,
                                 unsigned* __restrict__ dst) {
  const int gid  = blockIdx.x * blockDim.x + threadIdx.x;  // [0, 512*8*32)
  const int c    = gid >> 8;
  const int rem  = gid & 255;
  const int n    = rem >> 5;
  const int lane = rem & 31;
  const int z    = n * 16 + (lane & 15);
  const int kb   = c * 32 + ((lane >> 4) << 4);

  unsigned w[8];
#pragma unroll
  for (int e = 0; e < 8; ++e) {
    unsigned lo = f2bf(C[(size_t)(kb + 2 * e    ) * DVEC + z]);
    unsigned hi = f2bf(C[(size_t)(kb + 2 * e + 1) * DVEC + z]);
    w[e] = lo | (hi << 16);
  }
  unsigned* p = dst + (size_t)gid * 8;
#pragma unroll
  for (int e = 0; e < 8; ++e) p[e] = w[e];
}

// ---------------------------------------------------------------------------
// Kernel 2: Out(4096x128) = P(4096x16384) @ Cflat(16384x128) with on-the-fly
// rank-1 A fragments and v_wmma_f32_16x16x32_bf16. 4 waves/WG, 16 tokens/wave,
// each wave covers all 128 z (8 accumulator tiles).
// ---------------------------------------------------------------------------
__global__ void __launch_bounds__(128, 1)
cooc_wmma_kernel(const float* __restrict__ fa,
                 const v8u*  __restrict__ cp,
                 float*      __restrict__ out) {
  __shared__ float smem[64 * 256];   // 64 tokens x (128 func + 128 arg) fp32

  const int tid    = threadIdx.x;
  const int wave   = tid >> 5;
  const int lane   = tid & 31;
  const int row    = lane & 15;      // M row this lane serves in A tiles
  const int hi     = lane >> 4;      // K-half selector in this code's contiguous K order
  const int wg_tok = blockIdx.x * 64;

  // Stage this WG's 64 tokens (coalesced fp32).
  const float* src = fa + (size_t)wg_tok * 256;
  for (int idx = tid; idx < 64 * 256; idx += 128) smem[idx] = src[idx];
  __syncthreads();

  const int    tloc = wave * 16 + row;
  const float* frow = &smem[tloc * 256];       // func[row, 0..127]
  const float* arow = frow + 128;              // arg [row, 0..127]

  // arg values in A-fragment element order, for the 4 possible j0 = 32*s.
  // lanes 0-15 need K offsets {0..15}; lanes 16-31 need {16..31}.
  float af[4][16];
#pragma unroll
  for (int s = 0; s < 4; ++s) {
    const int bj = 32 * s + hi * 16;
#pragma unroll
    for (int e = 0; e < 8; ++e) {
      af[s][e]     = arow[bj + e];
      af[s][8 + e] = arow[bj + 8 + e];
    }
  }

  v8f acc[8] = {};

  for (int i = 0; i < 128; ++i) {
    const float f = frow[i];                   // func[row, i] (ds_load_b32)
#pragma unroll
    for (int s = 0; s < 4; ++s) {
      const int c = i * 4 + s;                 // K-chunk index

      // Keep the B stream hot: prefetch ~8 chunks (64 KB) ahead.
      __builtin_prefetch(&cp[(size_t)((c + 8) & (NCHUNK - 1)) * 256 + lane], 0, 1);

      // A fragment: rank-1 slice func[row,i] * arg[row, j0+K], fp32 -> bf16.
      v16f prod;
#pragma unroll
      for (int e = 0; e < 16; ++e) prod[e] = af[s][e] * f;
      v16bf a = __builtin_convertvector(prod, v16bf);

#pragma unroll
      for (int n = 0; n < 8; ++n) {
        v8u  braw = cp[(size_t)c * 256 + n * 32 + lane];  // 32B/lane from L2
        v16bf b   = __builtin_bit_cast(v16bf, braw);
        acc[n] = __builtin_amdgcn_wmma_f32_16x16x32_bf16(
            false, a, false, b, (short)0, acc[n], false, false);
      }
    }
  }

  // C/D layout: VGPR v, lane -> M = v + 8*hi, N = lane%16. Coalesced stores.
  float* orow = out + (size_t)(wg_tok + wave * 16) * DVEC;
#pragma unroll
  for (int n = 0; n < 8; ++n) {
#pragma unroll
    for (int v = 0; v < 8; ++v) {
      orow[(size_t)(v + hi * 8) * DVEC + n * 16 + (lane & 15)] = acc[n][v];
    }
  }
}

// ---------------------------------------------------------------------------
// Fallback (only if workspace is unexpectedly small): plain fp32 FMA version.
// ---------------------------------------------------------------------------
__global__ void cooc_fallback_kernel(const float* __restrict__ fa,
                                     const float* __restrict__ C,
                                     float* __restrict__ out) {
  const int t = blockIdx.x;
  const int z = threadIdx.x;
  const float* func = fa + (size_t)t * 256;
  const float* arg  = func + 128;
  float acc = 0.f;
  for (int i = 0; i < 128; ++i) {
    const float* Ci = C + (size_t)i * DVEC * DVEC + z;
    float inner = 0.f;
    for (int j = 0; j < 128; ++j)
      inner = __builtin_fmaf(arg[j], Ci[(size_t)j * DVEC], inner);
    acc = __builtin_fmaf(func[i], inner, acc);
  }
  out[(size_t)t * DVEC + z] = acc;
}

extern "C" void kernel_launch(void* const* d_in, const int* in_sizes, int n_in,
                              void* d_out, int out_size, void* d_ws, size_t ws_size,
                              hipStream_t stream) {
  const float* fa  = (const float*)d_in[0];   // (4,1024,256) fp32
  const float* C   = (const float*)d_in[1];   // (128,128,128) fp32
  float*       out = (float*)d_out;           // (4,1024,128) fp32
  (void)in_sizes; (void)n_in; (void)out_size;

  if (ws_size >= PACK_BYTES) {
    unsigned* cpack = (unsigned*)d_ws;
    cooc_pack_kernel<<<512, 256, 0, stream>>>(C, cpack);
    cooc_wmma_kernel<<<NTOK / 64, 128, 0, stream>>>(fa, (const v8u*)d_ws, out);
  } else {
    cooc_fallback_kernel<<<NTOK, DVEC, 0, stream>>>(fa, C, out);
  }
}
